// LinearAttention_60773787238668
// MI455X (gfx1250) — matrix operand order repaired and checked by reading, not compiled, not run on hardware
//
#include <hip/hip_runtime.h>
#include <hip/hip_bf16.h>
#include <stdint.h>

#define Bv 8
#define Nn 4096
#define Cc 512
#define Hh 64
#define Ww 64
#define NHEADS 16
#define HD 32

typedef __bf16 bf16;
typedef __attribute__((ext_vector_type(16))) __bf16 v16bf;
typedef __attribute__((ext_vector_type(8)))  __bf16 v8bf;
typedef __attribute__((ext_vector_type(4)))  __bf16 v4bf;
typedef __attribute__((ext_vector_type(8)))  float  v8f;
typedef unsigned int u32;
typedef __attribute__((ext_vector_type(4))) u32 u32x4;
typedef __attribute__((ext_vector_type(8))) u32 u32x8;

__device__ __forceinline__ v16bf concat8(v8bf lo, v8bf hi) {
  return __builtin_shufflevector(lo, hi, 0, 1, 2, 3, 4, 5, 6, 7, 8, 9, 10, 11, 12, 13, 14, 15);
}

// ---- gfx1250 async-copy helpers (GLOBAL_LOAD_ASYNC_TO_LDS_B128 / S_WAIT_ASYNCCNT) ----
__device__ __forceinline__ unsigned lds_addr32(const void* p) {
  // flat LDS address: low 32 bits are the LDS byte offset (ISA aperture rule)
  return (unsigned)(size_t)p;
}
__device__ __forceinline__ void async_load_b128(unsigned ldsaddr, const void* base, unsigned byteoff) {
  asm volatile("global_load_async_to_lds_b128 %0, %1, %2"
               :: "v"(ldsaddr), "v"(byteoff), "s"(base) : "memory");
}
__device__ __forceinline__ void wait_async0() {
  asm volatile("s_wait_asynccnt 0" ::: "memory");
}

// ---- gfx1250 Tensor Data Mover: 2D tile DMA via D# descriptor (ISA ch.8) ----
// group0: [0]=count=1, [1]=lds_addr, [2]=gaddr[31:0], [3]=gaddr[56:32]|type2<<30
// group1: data_size=4B; tensor_dim0/1, tile_dim0/1, tensor_dim0_stride per §8.4
__device__ __forceinline__ void tdm_load_2d_f32(unsigned lds_byte_addr, const void* gaddr,
                                                u32 tensor_d0, u32 tensor_d1,
                                                u32 tile_d0, u32 tile_d1, u32 stride0) {
  unsigned long long ga = (unsigned long long)(size_t)gaddr;
  u32x4 g0;
  g0[0] = 1u;                                   // count=1 (valid user descriptor)
  g0[1] = lds_byte_addr;                        // bits 63:32 lds_addr
  g0[2] = (u32)ga;                              // bits 95:64 global_addr lo
  g0[3] = (u32)(ga >> 32) | (2u << 30);         // global_addr[56:32] | type=2
  u32x8 g1;
  g1[0] = (2u << 16);                           // workgroup_mask=0, data_size=2 (4B)
  g1[1] = (tensor_d0 & 0xFFFFu) << 16;          // bits 63:48 = tensor_dim0[15:0]
  g1[2] = (tensor_d0 >> 16) | ((tensor_d1 & 0xFFFFu) << 16);  // dim0 hi | dim1 lo
  g1[3] = (tensor_d1 >> 16) | (tile_d0 << 16);  // dim1 hi | tile_dim0
  g1[4] = tile_d1 & 0xFFFFu;                    // tile_dim1 (tile_dim2 = 0)
  g1[5] = stride0;                              // tensor_dim0_stride[31:0]
  g1[6] = 0u;                                   // stride hi, dim1_stride lo
  g1[7] = 0u;
  asm volatile("tensor_load_to_lds %0, %1" :: "s"(g0), "s"(g1) : "memory");
}
__device__ __forceinline__ void wait_tensor0() {
  __builtin_amdgcn_s_wait_tensorcnt(0);
}

// ============================================================================
// Kernel 1: qk = x @ qk_w.T + qk_b  (M=32768, K=512, N=1024), transposed store.
// Block 256 thr = 8 waves; tile 64(M) x 128(N); waves on 4x2 grid, 16x64 each.
// A tile staged via async-to-LDS DMA; B tile staged via TDM (one 16-row slice
// per wave, descriptor in uniform SGPRs); f32->bf16 packed once per k-tile.
// ============================================================================
__global__ void qk_gemm(const float* __restrict__ x, const float* __restrict__ w,
                        const float* __restrict__ bias,
                        float* __restrict__ qn, float* __restrict__ kn) {
  __shared__ float Asf[64 * 32];    // 8 KB raw f32 A tile
  __shared__ float Bsf[128 * 32];   // 16 KB raw f32 B tile
  __shared__ bf16  Asb[64 * 32];    // 4 KB packed bf16
  __shared__ bf16  Bsb[128 * 32];   // 8 KB packed bf16
  const int tid  = threadIdx.x;
  const int lane = tid & 31, wave = tid >> 5;
  const int l15  = lane & 15;
  const int wm   = wave >> 1, wn = wave & 1;
  const int m0   = blockIdx.y * 64;
  const int n0   = blockIdx.x * 128;
  // wave index as a compiler-provable uniform (SGPR) for the TDM descriptor
  const int wvu  = __builtin_amdgcn_readfirstlane(tid) >> 5;

  v8f acc[4] = {};

  for (int kk = 0; kk < Cc; kk += 32) {
    __syncthreads();
    // ---- A tile: 64x32 f32 via async DMA to LDS (512 float4, 2 per thread) ----
    #pragma unroll
    for (int i = 0; i < 2; ++i) {
      int idx = tid + i * 256;
      int row = idx >> 3, seg = (idx & 7) * 4;
      async_load_b128(lds_addr32(&Asf[row * 32 + seg]), x,
                      (unsigned)(((m0 + row) * Cc + kk + seg) * 4));
    }
    // ---- B tile: 128x32 f32 via TDM; wave wvu DMAs rows [wvu*16, wvu*16+16) ----
    tdm_load_2d_f32(lds_addr32(&Bsf[wvu * 16 * 32]),
                    w + (size_t)(n0 + wvu * 16) * Cc + kk,
                    /*tensor_d0=*/Cc, /*tensor_d1=*/0x100000u,
                    /*tile_d0=*/32u, /*tile_d1=*/16u, /*stride0=*/Cc);
    wait_async0();
    wait_tensor0();
    __syncthreads();

    // ---- pack f32 -> bf16 once per tile (compiler emits v_cvt_pk_bf16_f32) ----
    #pragma unroll
    for (int i = 0; i < 2; ++i) {
      int q4 = tid + i * 256;                 // 512 float4 of A
      const float4 f = *(const float4*)&Asf[q4 * 4];
      v4bf o; o[0] = (bf16)f.x; o[1] = (bf16)f.y; o[2] = (bf16)f.z; o[3] = (bf16)f.w;
      *(v4bf*)&Asb[q4 * 4] = o;
    }
    #pragma unroll
    for (int i = 0; i < 4; ++i) {
      int q4 = tid + i * 256;                 // 1024 float4 of B
      const float4 f = *(const float4*)&Bsf[q4 * 4];
      v4bf o; o[0] = (bf16)f.x; o[1] = (bf16)f.y; o[2] = (bf16)f.z; o[3] = (bf16)f.w;
      *(v4bf*)&Bsb[q4 * 4] = o;
    }
    __syncthreads();

    // ---- fragments: pure ds_load_b128 pairs, zero per-WMMA conversion ----
    // A 16x32 bf16 layout: lanes<16 hold K 0..7 & 16..23, lanes>=16 K 8..15 & 24..31
    const int kbA = (lane < 16) ? 0 : 8;
    const bf16* ar = &Asb[(wm * 16 + l15) * 32];
    v16bf a = concat8(*(const v8bf*)(ar + kbA), *(const v8bf*)(ar + kbA + 16));

    const int kbB = (lane < 16) ? 0 : 8;
    #pragma unroll
    for (int t = 0; t < 4; ++t) {
      const bf16* br = &Bsb[(wn * 64 + t * 16 + l15) * 32];
      v16bf bb = concat8(*(const v8bf*)(br + kbB), *(const v8bf*)(br + kbB + 16));
      acc[t] = __builtin_amdgcn_wmma_f32_16x16x32_bf16(false, a, false, bb,
                                                       (short)0, acc[t], false, false);
    }
  }

  // store transposed into (B, C, N) buffers (q = first 512 cols, k = last 512)
  const int mrow0 = m0 + wm * 16 + ((lane < 16) ? 0 : 8);
  #pragma unroll
  for (int t = 0; t < 4; ++t) {
    int col = n0 + wn * 64 + t * 16 + l15;
    float bvv = bias[col];
    #pragma unroll
    for (int r = 0; r < 8; ++r) {
      int row = mrow0 + r;
      int b = row >> 12, n = row & (Nn - 1);
      float v = acc[t][r] + bvv;
      if (col < Cc) qn[((size_t)b * Cc + col) * Nn + n] = v;
      else          kn[((size_t)b * Cc + (col - Cc)) * Nn + n] = v;
    }
  }
}

// ============================================================================
// Kernel 2: dwconv3x3 + BN + ReLU + per-pixel sigmoid gate; pooled sums
// grid (H, B); block 256 = 64 pixels x 4 channel-strides
// ============================================================================
__global__ void spatial_op(const float* __restrict__ src, float* __restrict__ dst,
                           float* __restrict__ pool,
                           const float* __restrict__ dw_w, const float* __restrict__ dw_b,
                           const float* __restrict__ bn_g, const float* __restrict__ bn_b,
                           const float* __restrict__ pw_w) {
  __shared__ float sgate[Ww];
  const int tid = threadIdx.x;
  const int h = blockIdx.x, b = blockIdx.y;
  const int wp = tid & 63, cs = tid >> 6;
  const float bnscale = 0.9999950000374997f;  // 1/sqrt(1+1e-5)
  if (tid < Ww) sgate[tid] = 0.0f;
  __syncthreads();
  const size_t base_b = (size_t)b * Cc * Nn;
  float gpart = 0.0f;
  for (int c = cs; c < Cc; c += 4) {
    const float* sp = src + base_b + (size_t)c * Nn;
    float acc = dw_b[c];
    #pragma unroll
    for (int dh = -1; dh <= 1; ++dh) {
      int hh = h + dh;
      if (hh < 0 || hh >= Hh) continue;
      #pragma unroll
      for (int dw = -1; dw <= 1; ++dw) {
        int ww2 = wp + dw;
        if (ww2 < 0 || ww2 >= Ww) continue;
        acc += sp[hh * Ww + ww2] * dw_w[c * 9 + (dh + 1) * 3 + (dw + 1)];
      }
    }
    float y = acc * (bn_g[c] * bnscale) + bn_b[c];
    y = fmaxf(y, 0.0f);
    gpart += y * pw_w[c];
  }
  atomicAdd(&sgate[wp], gpart);   // ds_add_f32
  __syncthreads();
  const float s = 1.0f / (1.0f + __expf(-sgate[wp]));
  for (int c = cs; c < Cc; c += 4) {
    size_t idx = base_b + (size_t)c * Nn + h * Ww + wp;
    float v = src[idx] * s;
    dst[idx] = v;
    atomicAdd(&pool[b * Cc + c], v);
  }
}

// ============================================================================
// Kernel 3: channel gate = sigmoid(mean_pool @ w.T)   (8 x 512 x 512)
// ============================================================================
__global__ void channel_gate(const float* __restrict__ pool, const float* __restrict__ w,
                             float* __restrict__ gate) {
  __shared__ float ps[Cc];
  const int b = blockIdx.y;
  const int o = blockIdx.x * 256 + threadIdx.x;
  for (int i = threadIdx.x; i < Cc; i += 256) ps[i] = pool[b * Cc + i] * (1.0f / (float)Nn);
  __syncthreads();
  float acc = 0.0f;
  for (int c = 0; c < Cc; ++c) acc += ps[c] * w[o * Cc + c];
  gate[b * Cc + o] = 1.0f / (1.0f + __expf(-acc));
}

// ============================================================================
// Kernel 4: channel gate * (elu+1) * RoPE -> bf16; q:(b,h,n,d)  k:(b,h,d,n)+kmean
// ============================================================================
__global__ void finalize_qk(const float* __restrict__ gsp, const float* __restrict__ gate,
                            const float* __restrict__ rc, const float* __restrict__ rs,
                            bf16* __restrict__ outp, float* __restrict__ kmean_sum,
                            int transposed) {
  const int n = blockIdx.x, b = blockIdx.y;
  const int j = threadIdx.x;        // pair index 0..255
  const int c0 = 2 * j, c1 = 2 * j + 1;
  const size_t base = (size_t)b * Cc * Nn;
  float xr = gsp[base + (size_t)c0 * Nn + n] * gate[b * Cc + c0];
  float xi = gsp[base + (size_t)c1 * Nn + n] * gate[b * Cc + c1];
  xr = (xr > 0.0f) ? (xr + 1.0f) : __expf(xr);   // elu(x)+1
  xi = (xi > 0.0f) ? (xi + 1.0f) : __expf(xi);
  float co = rc[(size_t)n * 256 + j], si = rs[(size_t)n * 256 + j];
  float re = co * xr - si * xi;
  float im = si * xr + co * xi;
  int h0 = c0 >> 5, d0 = c0 & 31;
  int h1 = c1 >> 5, d1 = c1 & 31;
  if (transposed) {
    outp[(((size_t)b * NHEADS + h0) * HD + d0) * Nn + n] = (bf16)re;
    outp[(((size_t)b * NHEADS + h1) * HD + d1) * Nn + n] = (bf16)im;
  } else {
    outp[(((size_t)b * NHEADS + h0) * Nn + n) * HD + d0] = (bf16)re;
    outp[(((size_t)b * NHEADS + h1) * Nn + n) * HD + d1] = (bf16)im;
  }
  if (kmean_sum) {
    atomicAdd(&kmean_sum[(b * NHEADS + h0) * HD + d0], re);
    atomicAdd(&kmean_sum[(b * NHEADS + h1) * HD + d1], im);
  }
}

// v = x reshaped to (b, h, n, d), bf16
__global__ void vconv(const float* __restrict__ x, bf16* __restrict__ v) {
  const int n = blockIdx.x, b = blockIdx.y;
  #pragma unroll
  for (int i = 0; i < 2; ++i) {
    int c = threadIdx.x + i * 256;
    int h = c >> 5, d = c & 31;
    v[(((size_t)b * NHEADS + h) * Nn + n) * HD + d] = (bf16)x[((size_t)b * Nn + n) * Cc + c];
  }
}

// ============================================================================
// Kernel 5: kv[b,h] = K^T V / N  (32 x 4096 x 32) via WMMA, cross-wave LDS reduce
// ============================================================================
__global__ void kv_gemm(const bf16* __restrict__ kt, const bf16* __restrict__ vv,
                        float* __restrict__ kvout) {
  __shared__ float red[8 * 256];
  const int bh = blockIdx.x;
  const int tid = threadIdx.x, lane = tid & 31, wave = tid >> 5;
  const int l15 = lane & 15;
  const int kbA = (lane < 16) ? 0 : 16;
  const int kbB = (lane < 16) ? 0 : 16;
  const bf16* kbase = kt + (size_t)bh * HD * Nn;
  const bf16* vbase = vv + (size_t)bh * Nn * HD;
  v8f acc[2][2] = {};
  for (int ni = 0; ni < 16; ++ni) {
    const int nb = wave * 512 + ni * 32;
    v16bf a[2], bb[2];
    #pragma unroll
    for (int mt = 0; mt < 2; ++mt) {
      const bf16* kr = kbase + (size_t)(mt * 16 + l15) * Nn + nb;
      a[mt] = concat8(*(const v8bf*)(kr + kbA), *(const v8bf*)(kr + kbA + 8));
    }
    #pragma unroll
    for (int nt = 0; nt < 2; ++nt) {
      const int e = nt * 16 + l15;
      #pragma unroll
      for (int j = 0; j < 16; ++j) bb[nt][j] = vbase[(size_t)(nb + kbB + j) * HD + e];
    }
    #pragma unroll
    for (int mt = 0; mt < 2; ++mt)
      #pragma unroll
      for (int nt = 0; nt < 2; ++nt)
        acc[mt][nt] = __builtin_amdgcn_wmma_f32_16x16x32_bf16(false, a[mt], false, bb[nt],
                                                              (short)0, acc[mt][nt], false, false);
  }
  const int mbase = (lane < 16) ? 0 : 8;
  for (int mt = 0; mt < 2; ++mt) {
    for (int nt = 0; nt < 2; ++nt) {
      __syncthreads();
      #pragma unroll
      for (int r = 0; r < 8; ++r) red[wave * 256 + (mbase + r) * 16 + l15] = acc[mt][nt][r];
      __syncthreads();
      float s = 0.0f;
      #pragma unroll
      for (int wv2 = 0; wv2 < 8; ++wv2) s += red[wv2 * 256 + tid];
      int d = mt * 16 + (tid >> 4), e = nt * 16 + (tid & 15);
      kvout[(size_t)bh * HD * HD + d * HD + e] = s * (1.0f / (float)Nn);
    }
  }
}

// ============================================================================
// Kernel 6: out = (Q @ kv) * z,  z = 1/(Q . kmean + 1e-6)
// ============================================================================
__global__ void out_gemm(const bf16* __restrict__ q, const float* __restrict__ kvbuf,
                         const float* __restrict__ kmean_sum, float* __restrict__ out) {
  __shared__ bf16 kvs[HD * HD];
  __shared__ float kms[HD];
  __shared__ float zs[8 * 16];
  const int tid = threadIdx.x, lane = tid & 31, wave = tid >> 5;
  const int l15 = lane & 15;
  const int head = blockIdx.y, b = blockIdx.z;
  const int bh = b * NHEADS + head;
  for (int i = tid; i < HD * HD; i += 256) kvs[i] = (bf16)kvbuf[(size_t)bh * HD * HD + i];
  if (tid < HD) kms[tid] = kmean_sum[bh * HD + tid] * (1.0f / (float)Nn);
  __syncthreads();

  const int r0 = blockIdx.x * 128 + wave * 16;
  const bf16* qb = q + ((size_t)bh * Nn + r0) * HD;
  const int kbA = (lane < 16) ? 0 : 16;
  const int kbB = (lane < 16) ? 0 : 16;
  const bf16* qr0 = qb + (size_t)l15 * HD;
  v16bf a = concat8(*(const v8bf*)(qr0 + kbA), *(const v8bf*)(qr0 + kbA + 8));
  v8f acc[2] = {};
  #pragma unroll
  for (int nt = 0; nt < 2; ++nt) {
    v16bf bb;
    #pragma unroll
    for (int j = 0; j < 16; ++j) bb[j] = kvs[(kbB + j) * HD + nt * 16 + l15];
    acc[nt] = __builtin_amdgcn_wmma_f32_16x16x32_bf16(false, a, false, bb,
                                                      (short)0, acc[nt], false, false);
  }
  if (lane < 16) {
    const bf16* qr = qb + (size_t)lane * HD;
    float dot = 0.0f;
    #pragma unroll
    for (int d = 0; d < HD; ++d) dot += (float)qr[d] * kms[d];
    zs[wave * 16 + lane] = 1.0f / (dot + 1e-6f);
  }
  __syncthreads();
  const int mbase = (lane < 16) ? 0 : 8;
  #pragma unroll
  for (int nt = 0; nt < 2; ++nt) {
    int c = head * HD + nt * 16 + l15;
    #pragma unroll
    for (int r = 0; r < 8; ++r) {
      int m = mbase + r;
      int row = r0 + m;
      out[((size_t)b * Nn + row) * Cc + c] = acc[nt][r] * zs[wave * 16 + m];
    }
  }
}

// Kernel 7: out += dwconv3x3(x) + lepe_b
__global__ void lepe_add(const float* __restrict__ x, const float* __restrict__ lw,
                         const float* __restrict__ lb, float* __restrict__ out) {
  const int n = blockIdx.x, b = blockIdx.z;
  const int c = blockIdx.y * 256 + threadIdx.x;
  const int h = n >> 6, w = n & 63;
  float acc = lb[c];
  #pragma unroll
  for (int dh = -1; dh <= 1; ++dh) {
    int hh = h + dh; if (hh < 0 || hh >= Hh) continue;
    #pragma unroll
    for (int dw = -1; dw <= 1; ++dw) {
      int ww2 = w + dw; if (ww2 < 0 || ww2 >= Ww) continue;
      acc += x[((size_t)b * Nn + (hh * Ww + ww2)) * Cc + c] * lw[c * 9 + (dh + 1) * 3 + (dw + 1)];
    }
  }
  out[((size_t)b * Nn + n) * Cc + c] += acc;
}

__global__ void zero_init(float* __restrict__ p, int count) {
  int i = blockIdx.x * 256 + threadIdx.x;
  if (i < count) p[i] = 0.0f;
}

extern "C" void kernel_launch(void* const* d_in, const int* in_sizes, int n_in,
                              void* d_out, int out_size, void* d_ws, size_t ws_size,
                              hipStream_t stream) {
  const float* x        = (const float*)d_in[0];
  const float* qk_w     = (const float*)d_in[1];
  const float* qk_b     = (const float*)d_in[2];
  const float* qsp_dw_w = (const float*)d_in[3];
  const float* qsp_dw_b = (const float*)d_in[4];
  const float* qsp_bn_g = (const float*)d_in[5];
  const float* qsp_bn_b = (const float*)d_in[6];
  const float* qsp_pw_w = (const float*)d_in[7];
  const float* qch_w    = (const float*)d_in[8];
  const float* ksp_dw_w = (const float*)d_in[9];
  const float* ksp_dw_b = (const float*)d_in[10];
  const float* ksp_bn_g = (const float*)d_in[11];
  const float* ksp_bn_b = (const float*)d_in[12];
  const float* ksp_pw_w = (const float*)d_in[13];
  const float* kch_w    = (const float*)d_in[14];
  const float* lepe_w   = (const float*)d_in[15];
  const float* lepe_b   = (const float*)d_in[16];
  const float* rope_cos = (const float*)d_in[17];
  const float* rope_sin = (const float*)d_in[18];
  float* out = (float*)d_out;

  char* ws = (char*)d_ws;
  const size_t SZ = (size_t)Bv * Cc * Nn * sizeof(float);       // 64 MB
  float* q_nchw = (float*)(ws);
  float* k_nchw = (float*)(ws + SZ);
  float* q_gsp  = (float*)(ws + 2 * SZ);
  float* k_gsp  = (float*)(ws + 3 * SZ);
  float* pool_q = (float*)(ws + 4 * SZ);
  float* pool_k = pool_q + Bv * Cc;
  float* kmean  = pool_k + Bv * Cc;          // B*HEADS*HD
  float* gate_q = kmean + Bv * NHEADS * HD;
  float* gate_k = gate_q + Bv * Cc;
  float* kvbuf  = gate_k + Bv * Cc;          // B*HEADS*HD*HD
  bf16*  q_bf   = (bf16*)(kvbuf + Bv * NHEADS * HD * HD);
  bf16*  kt_bf  = q_bf + (size_t)Bv * NHEADS * Nn * HD;
  bf16*  v_bf   = kt_bf + (size_t)Bv * NHEADS * Nn * HD;

  const int nzero = Bv * Cc * 2 + Bv * NHEADS * HD;  // pool_q, pool_k, kmean (contiguous)
  zero_init<<<dim3((nzero + 255) / 256), 256, 0, stream>>>(pool_q, nzero);

  qk_gemm<<<dim3(8, 512), 256, 0, stream>>>(x, qk_w, qk_b, q_nchw, k_nchw);

  spatial_op<<<dim3(Hh, Bv), 256, 0, stream>>>(q_nchw, q_gsp, pool_q,
      qsp_dw_w, qsp_dw_b, qsp_bn_g, qsp_bn_b, qsp_pw_w);
  spatial_op<<<dim3(Hh, Bv), 256, 0, stream>>>(k_nchw, k_gsp, pool_k,
      ksp_dw_w, ksp_dw_b, ksp_bn_g, ksp_bn_b, ksp_pw_w);

  channel_gate<<<dim3(2, Bv), 256, 0, stream>>>(pool_q, qch_w, gate_q);
  channel_gate<<<dim3(2, Bv), 256, 0, stream>>>(pool_k, kch_w, gate_k);

  finalize_qk<<<dim3(Nn, Bv), 256, 0, stream>>>(q_gsp, gate_q, rope_cos, rope_sin,
                                                q_bf, nullptr, 0);
  finalize_qk<<<dim3(Nn, Bv), 256, 0, stream>>>(k_gsp, gate_k, rope_cos, rope_sin,
                                                kt_bf, kmean, 1);
  vconv<<<dim3(Nn, Bv), 256, 0, stream>>>(x, v_bf);

  kv_gemm<<<dim3(Bv * NHEADS), 256, 0, stream>>>(kt_bf, v_bf, kvbuf);
  out_gemm<<<dim3(Nn / 128, NHEADS, Bv), 256, 0, stream>>>(q_bf, kvbuf, kmean, out);
  lepe_add<<<dim3(Nn, 2, Bv), 256, 0, stream>>>(x, lepe_w, lepe_b, out);
}
